// _HFInternLM3Attention_18485539242431
// MI455X (gfx1250) — matrix operand order repaired and checked by reading, not compiled, not run on hardware
//
#include <hip/hip_runtime.h>
#include <hip/hip_bf16.h>
#include <cstdint>

// ---------------- problem constants ----------------
constexpr int S   = 4096;
constexpr int H   = 2048;
constexpr int NH  = 16;
constexpr int NKV = 4;
constexpr int HD  = 128;
constexpr int NREP = NH / NKV;
constexpr float QSCALE = 0.08838834764831845f; // 1/sqrt(128)

// ---------------- WMMA vector types ----------------
typedef __attribute__((ext_vector_type(16))) __bf16 v16bf;
typedef __attribute__((ext_vector_type(8)))  __bf16 v8bf;
typedef __attribute__((ext_vector_type(8)))  float  v8f;

__device__ __forceinline__ v16bf ld2x8(const __bf16* p0, const __bf16* p1) {
  union { v16bf v; v8bf h[2]; } u;
  u.h[0] = *(const v8bf*)p0;
  u.h[1] = *(const v8bf*)p1;
  return u.v;
}

__device__ __forceinline__ v8f wmma_bf16(v16bf a, v16bf b, v8f c) {
  return __builtin_amdgcn_wmma_f32_16x16x32_bf16(
      /*neg_a=*/false, a, /*neg_b=*/false, b,
      /*c_mod=*/(short)0, c, /*reuse_a=*/false, /*reuse_b=*/false);
}

// CDNA5 async copy: global -> LDS, 16 bytes per lane, tracked by ASYNCcnt.
// lds_off is the wave-relative LDS byte address (group segment starts at 0).
__device__ __forceinline__ void async_g2l_b128(const __bf16* g, unsigned lds_off) {
  asm volatile("global_load_async_to_lds_b128 %0, %1, off"
               :: "v"(lds_off), "v"((unsigned long long)(size_t)g)
               : "memory");
}
__device__ __forceinline__ void wait_asynccnt0() {
  asm volatile("s_wait_asynccnt 0x0" ::: "memory");
}

// 16-lane butterfly reductions (wave32: masks 1,2,4,8 stay within each half)
__device__ __forceinline__ float rmax16(float x) {
  x = fmaxf(x, __shfl_xor(x, 1, 32));
  x = fmaxf(x, __shfl_xor(x, 2, 32));
  x = fmaxf(x, __shfl_xor(x, 4, 32));
  x = fmaxf(x, __shfl_xor(x, 8, 32));
  return x;
}
__device__ __forceinline__ float rsum16(float x) {
  x += __shfl_xor(x, 1, 32);
  x += __shfl_xor(x, 2, 32);
  x += __shfl_xor(x, 4, 32);
  x += __shfl_xor(x, 8, 32);
  return x;
}

// ---------------- f32 -> bf16 conversion ----------------
__global__ void cvt_bf16(const float* __restrict__ in, __bf16* __restrict__ out, int n) {
  int i = blockIdx.x * blockDim.x + threadIdx.x;
  if (i < n) out[i] = (__bf16)in[i];
}

// ---------------- tiled WMMA GEMM: C[M,N] = A[M,K] @ B[K,N] ----------------
// block = 256 threads (8 waves), tile 128x128, K-step 32, double-buffered LDS,
// A tile staged with CDNA5 async global->LDS copies (overlaps the WMMAs).
// M,N multiples of 128; K multiple of 32.
__global__ __launch_bounds__(256)
void gemm_bf16(const __bf16* __restrict__ A, const __bf16* __restrict__ B,
               float* __restrict__ C, int M, int N, int K) {
  // layout: As[0]=smem[0..4095], As[1]=smem[4096..], Bs[0]=smem[8192..], Bs[1]=smem[12288..]
  __shared__ __bf16 smem[16384];   // 32 KB

  const int tid  = threadIdx.x;
  const int lane = tid & 31;
  const int wid  = tid >> 5;
  const int lm   = lane & 15;
  const int hi   = lane >> 4;

  const int waveM = wid & 1;        // 2 waves over M
  const int waveN = wid >> 1;       // 4 waves over N
  const int m_base = waveM * 64;
  const int n_base = waveN * 32;

  const int blockM = blockIdx.y * 128;
  const int blockN = blockIdx.x * 128;

  // staging coordinates (per thread)
  const int arow  = tid >> 1;       // 0..127
  const int ahalf = tid & 1;        // 0..1  (16 bf16 each)
  const int bn0   = (tid & 15) * 8; // 0..120
  const int bk    = tid >> 4;       // 0..15

  // issue async A staging + transposed B staging into buffer `buf`
  auto stageA = [&](int buf, int k0) {
    const __bf16* ga = A + (size_t)(blockM + arow) * K + k0 + ahalf * 16;
    const unsigned lo = (unsigned)((buf * 4096 + arow * 32 + ahalf * 16) * 2);
    async_g2l_b128(ga,     lo);
    async_g2l_b128(ga + 8, lo + 16);
  };
  auto stageB = [&](int buf, int k0) {
    __bf16* Bsb = smem + 8192 + buf * 4096;  // [n][k] transposed
    union { uint4 u; __bf16 b[8]; } q0, q1;
    q0.u = *(const uint4*)(B + (size_t)(k0 + bk) * N + blockN + bn0);
    q1.u = *(const uint4*)(B + (size_t)(k0 + bk + 16) * N + blockN + bn0);
    #pragma unroll
    for (int j = 0; j < 8; ++j) {
      Bsb[(bn0 + j) * 32 + bk]      = q0.b[j];
      Bsb[(bn0 + j) * 32 + bk + 16] = q1.b[j];
    }
  };

  v8f acc[4][2];
  #pragma unroll
  for (int mf = 0; mf < 4; ++mf)
    #pragma unroll
    for (int nf = 0; nf < 2; ++nf)
      acc[mf][nf] = (v8f){0.f,0.f,0.f,0.f,0.f,0.f,0.f,0.f};

  const int nIter = K / 32;
  stageA(0, 0);
  stageB(0, 0);

  for (int it = 0; it < nIter; ++it) {
    const int cur = it & 1;
    const int nxt = cur ^ 1;
    const int k0  = it * 32;

    wait_asynccnt0();     // my async copies into `cur` are done
    __syncthreads();      // everyone's copies (async + ds stores) visible

    if (it + 1 < nIter) { // stage next tile; copies overlap the WMMAs below
      stageA(nxt, k0 + 32);
      stageB(nxt, k0 + 32);
      if (it + 2 < nIter) {
        __builtin_prefetch(A + (size_t)(blockM + arow) * K + k0 + 64, 0, 0);
        __builtin_prefetch(B + (size_t)(k0 + 32 + bk) * N + blockN + bn0, 0, 0);
      }
    }

    const __bf16* Asb = smem + cur * 4096;
    const __bf16* Bsb = smem + 8192 + cur * 4096;

    // A fragments (16x32 bf16): lane m, 16 contiguous k at 16*hi
    v16bf af[4];
    #pragma unroll
    for (int mf = 0; mf < 4; ++mf) {
      const __bf16* ar = &Asb[(m_base + mf * 16 + lm) * 32];
      af[mf] = ld2x8(ar + 16 * hi, ar + 16 * hi + 8);
    }
    // B fragments (32x16 bf16 B-layout): lane n, 16 contiguous k at 16*hi
    v16bf bf_[2];
    #pragma unroll
    for (int nf = 0; nf < 2; ++nf) {
      const __bf16* br = &Bsb[(n_base + nf * 16 + lm) * 32 + 16 * hi];
      bf_[nf] = ld2x8(br, br + 8);
    }
    #pragma unroll
    for (int mf = 0; mf < 4; ++mf)
      #pragma unroll
      for (int nf = 0; nf < 2; ++nf)
        acc[mf][nf] = wmma_bf16(af[mf], bf_[nf], acc[mf][nf]);

    __syncthreads();      // protect `nxt`-buffer reuse two iterations out
  }

  // C layout: VGPR r -> row r + 8*hi, lane[0:15] -> col
  #pragma unroll
  for (int mf = 0; mf < 4; ++mf)
    #pragma unroll
    for (int nf = 0; nf < 2; ++nf) {
      const int col = blockN + n_base + nf * 16 + lm;
      #pragma unroll
      for (int r = 0; r < 8; ++r) {
        const int row = blockM + m_base + mf * 16 + r + 8 * hi;
        C[(size_t)row * N + col] = acc[mf][nf][r];
      }
    }
}

// ---------------- RoPE + layout pack ----------------
// mode 0: q f32[s][h][d] -> rope -> bf16 Qp[h][s][d] * QSCALE
// mode 1: k f32[s][g][d] -> rope -> bf16 Kp[g][s][d]
// mode 2: v f32[s][g][d] ->          bf16 VT[g][d][s]
__global__ void rope_pack(const float* __restrict__ src,
                          const float* __restrict__ cs,
                          const float* __restrict__ sn,
                          __bf16* __restrict__ dst,
                          int heads, int mode, float scale) {
  const int idx = blockIdx.x * blockDim.x + threadIdx.x;
  const int total = S * heads * HD;
  if (idx >= total) return;
  const int d = idx % HD;
  const int t = idx / HD;
  const int h = t % heads;
  const int s = t / heads;
  const int W = heads * HD;
  const size_t soff = (size_t)s * W + h * HD;
  if (mode == 2) {
    dst[((size_t)h * HD + d) * S + s] = (__bf16)src[soff + d];
    return;
  }
  const float x     = src[soff + d];
  const float other = src[soff + (d < 64 ? d + 64 : d - 64)];
  const float c = cs[s * HD + d];
  const float si = sn[s * HD + d];
  const float r = x * c + (d < 64 ? -other : other) * si;
  dst[((size_t)h * S + s) * HD + d] = (__bf16)(r * scale);
}

// ---------------- flash attention (wave-level, causal, GQA) ----------------
// grid: NH*(S/16)/4 blocks of 128 threads; each wave owns one (head, 16-row q tile).
__global__ __launch_bounds__(128)
void attn_fwd(const __bf16* __restrict__ Qp,   // [NH][S][HD]
              const __bf16* __restrict__ Kp,   // [NKV][S][HD]
              const __bf16* __restrict__ VT,   // [NKV][HD][S]
              __bf16* __restrict__ AO) {       // [S][NH*HD]
  __shared__ __bf16 ldsP[4][16][32];           // per-wave P transpose tile

  const int lane = threadIdx.x & 31;
  const int wid  = threadIdx.x >> 5;
  const int lm   = lane & 15;
  const int hi   = lane >> 4;

  const int gw   = blockIdx.x * 4 + wid;
  const int h    = gw / (S / 16);
  const int q0   = (gw % (S / 16)) * 16;
  const int g    = h / NREP;

  // Q fragments: 4 chunks of 16x32 over HD=128 (A-layout, scale pre-folded)
  v16bf qa[4];
  {
    const __bf16* qb = Qp + ((size_t)h * S + q0 + lm) * HD;
    #pragma unroll
    for (int c = 0; c < 4; ++c)
      qa[c] = ld2x8(qb + 32 * c + 16 * hi, qb + 32 * c + 16 * hi + 8);
  }

  v8f o[8];
  #pragma unroll
  for (int f = 0; f < 8; ++f) o[f] = (v8f){0.f,0.f,0.f,0.f,0.f,0.f,0.f,0.f};
  float m_s[8], l_s[8];
  #pragma unroll
  for (int r = 0; r < 8; ++r) { m_s[r] = -1e30f; l_s[r] = 0.f; }

  const __bf16* kbase = Kp + (size_t)g * S * HD;
  const __bf16* vbase = VT + (size_t)g * HD * S;
  const int kt_max = (q0 + 15) / 32;   // wave-uniform

  for (int kt = 0; kt <= kt_max; ++kt) {
    // ---- scores: two 16x16 frags over a 32-key tile ----
    v8f s0 = (v8f){0.f,0.f,0.f,0.f,0.f,0.f,0.f,0.f};
    v8f s1 = s0;
    const __bf16* k0p = kbase + (size_t)(kt * 32 + lm) * HD + 16 * hi;
    const __bf16* k1p = k0p + (size_t)16 * HD;
    if (kt < kt_max) {                 // prefetch next 32-key tile (K and V)
      __builtin_prefetch(k0p + (size_t)32 * HD, 0, 0);
      __builtin_prefetch(k1p + (size_t)32 * HD, 0, 0);
      __builtin_prefetch(vbase + (size_t)lm * S + kt * 32 + 32, 0, 0);
    }
    #pragma unroll
    for (int c = 0; c < 4; ++c) {
      v16bf kb0 = ld2x8(k0p + 32 * c, k0p + 32 * c + 8);
      v16bf kb1 = ld2x8(k1p + 32 * c, k1p + 32 * c + 8);
      s0 = wmma_bf16(qa[c], kb0, s0);
      s1 = wmma_bf16(qa[c], kb1, s1);
    }

    // ---- online softmax (causal mask applied branchlessly) ----
    float a8[8];
    #pragma unroll
    for (int r = 0; r < 8; ++r) {
      const int row  = q0 + r + 8 * hi;
      const int key0 = kt * 32 + lm;
      const int key1 = key0 + 16;
      const float v0 = (key0 <= row) ? s0[r] : -1e30f;
      const float v1 = (key1 <= row) ? s1[r] : -1e30f;
      float mt = rmax16(fmaxf(v0, v1));
      const float mn = fmaxf(m_s[r], mt);
      const float alpha = __expf(m_s[r] - mn);
      m_s[r] = mn;
      const float p0 = __expf(v0 - mn);
      const float p1 = __expf(v1 - mn);
      l_s[r] = l_s[r] * alpha + rsum16(p0 + p1);
      a8[r] = alpha;
      ldsP[wid][r + 8 * hi][lm]      = (__bf16)p0;
      ldsP[wid][r + 8 * hi][16 + lm] = (__bf16)p1;
    }
    #pragma unroll
    for (int f = 0; f < 8; ++f)
      #pragma unroll
      for (int r = 0; r < 8; ++r)
        o[f][r] *= a8[r];

    // wait for wave-private LDS stores (cross-lane transpose), CDNA5 split counter
    asm volatile("s_wait_dscnt 0x0" ::: "memory");

    // reload P in A-layout (16x32)
    const __bf16* pr = &ldsP[wid][lm][0];
    v16bf pa = ld2x8(pr + 16 * hi, pr + 16 * hi + 8);

    // ---- O += P @ V : 8 dim-frags, B = VT contiguous over keys ----
    #pragma unroll
    for (int f = 0; f < 8; ++f) {
      const __bf16* vp = vbase + (size_t)(f * 16 + lm) * S + kt * 32 + 16 * hi;
      v16bf vb = ld2x8(vp, vp + 8);
      o[f] = wmma_bf16(pa, vb, o[f]);
    }
  }

  // ---- epilogue: normalize + store bf16 into AO[s][h*128+d] ----
  #pragma unroll
  for (int f = 0; f < 8; ++f) {
    const int col = h * HD + f * 16 + lm;
    #pragma unroll
    for (int r = 0; r < 8; ++r) {
      const int row = q0 + r + 8 * hi;
      AO[(size_t)row * (NH * HD) + col] = (__bf16)(o[f][r] / l_s[r]);
    }
  }
}

// ---------------- launcher ----------------
extern "C" void kernel_launch(void* const* d_in, const int* in_sizes, int n_in,
                              void* d_out, int out_size, void* d_ws, size_t ws_size,
                              hipStream_t stream) {
  (void)in_sizes; (void)n_in; (void)out_size; (void)ws_size;
  const float* hs   = (const float*)d_in[0];
  const float* cosp = (const float*)d_in[1];
  const float* sinp = (const float*)d_in[2];
  const float* Wq   = (const float*)d_in[3];
  const float* Wk   = (const float*)d_in[4];
  const float* Wv   = (const float*)d_in[5];
  const float* Wo   = (const float*)d_in[6];
  float* out = (float*)d_out;

  char* w = (char*)d_ws;
  auto carve = [&](size_t bytes) {
    void* p = (void*)w;
    w += (bytes + 255) & ~(size_t)255;
    return p;
  };
  __bf16* Xb  = (__bf16*)carve((size_t)S * H * 2);
  __bf16* Wqb = (__bf16*)carve((size_t)H * NH * HD * 2);
  __bf16* Wkb = (__bf16*)carve((size_t)H * NKV * HD * 2);
  __bf16* Wvb = (__bf16*)carve((size_t)H * NKV * HD * 2);
  __bf16* Wob = (__bf16*)carve((size_t)NH * HD * H * 2);
  float*  qf  = (float*)carve((size_t)S * NH * HD * 4);
  float*  kf  = (float*)carve((size_t)S * NKV * HD * 4);
  float*  vf  = (float*)carve((size_t)S * NKV * HD * 4);
  __bf16* Qp  = (__bf16*)carve((size_t)NH * S * HD * 2);
  __bf16* Kp  = (__bf16*)carve((size_t)NKV * S * HD * 2);
  __bf16* VTp = (__bf16*)carve((size_t)NKV * HD * S * 2);
  __bf16* AO  = (__bf16*)qf;   // qf fully consumed before attention writes AO

  auto cvt = [&](const float* src, __bf16* dst, int n) {
    cvt_bf16<<<(n + 255) / 256, 256, 0, stream>>>(src, dst, n);
  };
  cvt(hs, Xb, S * H);
  cvt(Wq, Wqb, H * NH * HD);
  cvt(Wk, Wkb, H * NKV * HD);
  cvt(Wv, Wvb, H * NKV * HD);
  cvt(Wo, Wob, NH * HD * H);

  // QKV projections
  gemm_bf16<<<dim3((NH * HD) / 128, S / 128), 256, 0, stream>>>(Xb, Wqb, qf, S, NH * HD, H);
  gemm_bf16<<<dim3((NKV * HD) / 128, S / 128), 256, 0, stream>>>(Xb, Wkb, kf, S, NKV * HD, H);
  gemm_bf16<<<dim3((NKV * HD) / 128, S / 128), 256, 0, stream>>>(Xb, Wvb, vf, S, NKV * HD, H);

  // RoPE + pack (Q scaled by 1/sqrt(HD); V transposed to [g][d][s])
  rope_pack<<<(S * NH * HD) / 256, 256, 0, stream>>>(qf, cosp, sinp, Qp, NH, 0, QSCALE);
  rope_pack<<<(S * NKV * HD) / 256, 256, 0, stream>>>(kf, cosp, sinp, Kp, NKV, 1, 1.0f);
  rope_pack<<<(S * NKV * HD) / 256, 256, 0, stream>>>(vf, cosp, sinp, VTp, NKV, 2, 1.0f);

  // flash attention: NH * (S/16) waves, 4 waves/block
  attn_fwd<<<(NH * (S / 16)) / 4, 128, 0, stream>>>(Qp, Kp, VTp, AO);

  // output projection
  gemm_bf16<<<dim3(H / 128, S / 128), 256, 0, stream>>>(AO, Wob, out, S, H, H);
}
